// model_base_61916248539507
// MI455X (gfx1250) — matrix operand order repaired and checked by reading, not compiled, not run on hardware
//
#include <hip/hip_runtime.h>
#include <hip/hip_bf16.h>

// ---------------------------------------------------------------------------
// Diffusion conditioning pipeline for MI455X (gfx1250, wave32, WMMA).
// Heavy op: per-batch GEMM ue[200x160] @ w2^T[160x160] -> sigmoid -> score.
// v3: fast sigmoid via v_rcp_f32 (no IEEE div chain); mean/m1/uemb reduction
// phases split 2-way so 320/416 threads are active with half-length loops.
// ---------------------------------------------------------------------------

typedef __attribute__((ext_vector_type(16))) _Float16 v16h;
typedef __attribute__((ext_vector_type(8)))  float    v8f;

#define B_        2048
#define L_        200
#define LHALF     100
#define EMB       32
#define TDIM      128
#define LDIM      32
#define COND      160
#define MTILES    13
#define NTILES    10
#define KSTEPS    5            // 160 / 32
#define LPAD      (MTILES * 16)
#define UE_STRIDE 168          // f16 elems; 336B row stride -> conflict-free
#define W2_STRIDE 168
#define OUT_ROWS  321
#define MASK_TIME 48
#define NTHREADS  416          // 13 wave32 waves
#define LOG2_10000 13.287712379549449f

// ---------------------------------------------------------------------------
// Kernel 1: ST_info rows [0,160) and noisy row 320.
// ---------------------------------------------------------------------------
__global__ void st_noisy_kernel(const int* __restrict__ tp,
                                const int* __restrict__ loc,
                                const float* __restrict__ emb,
                                const float* __restrict__ loc_emb_all,
                                const float* __restrict__ noise,
                                const int* __restrict__ t,
                                const float* __restrict__ alpha_hat,
                                float* __restrict__ out) {
    __shared__ float side[COND];
    const int b = blockIdx.x;
    const int tid = threadIdx.x;

    if (tid < TDIM / 2) {
        // div_j = 10000^(-j/64); interleaved sin/cos (1-D branch: no mask)
        const float div = exp2f(-((float)tid * (1.0f / 64.0f)) * LOG2_10000);
        const float ang = (float)tp[b] * div;
        side[2 * tid]     = __sinf(ang);
        side[2 * tid + 1] = __cosf(ang);
    }
    if (tid >= 64 && tid < 64 + LDIM) {
        const int k = tid - 64;
        side[TDIM + k] = loc_emb_all[(size_t)(loc[b] - 1) * LDIM + k];
    }
    if (tid < EMB) {
        const float a  = alpha_hat[t[b]];
        const float nv = sqrtf(a) * emb[b * EMB + tid] +
                         sqrtf(1.0f - a) * noise[b * EMB + tid];
        out[(size_t)b * OUT_ROWS * EMB + 320 * EMB + tid] = nv;
    }
    __syncthreads();

    for (int i = tid; i < COND * EMB; i += blockDim.x) {
        const int c = i / EMB;
        const int e = i % EMB;
        out[(size_t)b * OUT_ROWS * EMB + c * EMB + e] = side[c];
    }
}

// ---------------------------------------------------------------------------
// Kernel 2: user_info rows [160, 320). One workgroup (13 waves) per batch.
// ---------------------------------------------------------------------------
__global__ void user_info_kernel(const int* __restrict__ ST,          // [B,L,2]
                                 const float* __restrict__ loc_emb_all,
                                 const float* __restrict__ w1,        // [COND,COND]
                                 const float* __restrict__ w2,        // [COND,COND]
                                 const float* __restrict__ b2,
                                 const float* __restrict__ wc,
                                 float* __restrict__ out) {
    extern __shared__ char smem[];
    _Float16* ueS    = (_Float16*)smem;                    // LPAD * UE_STRIDE
    _Float16* w2S    = ueS + LPAD * UE_STRIDE;             // COND * W2_STRIDE
    float*    scoreS = (float*)(w2S + COND * W2_STRIDE);   // LPAD
    float*    meanS  = scoreS + LPAD;                      // COND
    float*    m1S    = meanS + COND;                       // COND
    float*    wcS    = m1S + COND;                         // COND
    float*    divS   = wcS + COND;                         // 64
    float*    uembS  = divS + 64;                          // COND
    float*    partS  = uembS + COND;                       // 2 * COND

    const int b   = blockIdx.x;
    const int tid = threadIdx.x;           // 416 threads = 13 waves

    // -- Stage constants ---------------------------------------------------
    if (tid < 64)
        divS[tid] = exp2f(-((float)tid * (1.0f / 64.0f)) * LOG2_10000);
    if (tid < COND) wcS[tid] = wc[tid];

    // -- w2 -> LDS f16, [c][k] row-major so B K-pairs are contiguous ------
    for (int i = tid; i < COND * COND; i += NTHREADS) {
        const int c = i / COND, k = i % COND;
        w2S[c * W2_STRIDE + k] = (_Float16)w2[i];
    }
    __syncthreads();

    // -- Build ue[l][k] in f16: k<32 loc gather, k>=32 masked sinusoid ----
    for (int i = tid; i < L_ * COND; i += NTHREADS) {
        const int l  = i / COND, k = i % COND;
        const int sI = ST[((size_t)b * L_ + l) * 2 + 0];
        const int ts = ST[((size_t)b * L_ + l) * 2 + 1];
        float v;
        if (k < LDIM) {
            v = loc_emb_all[(size_t)(sI - 1) * LDIM + k];
        } else if (ts == MASK_TIME) {
            v = 0.0f;                       // pe[pos == 48] = 0 (2-D branch)
        } else {
            const int j = k - LDIM;
            const float ang = (float)ts * divS[j >> 1];
            v = (j & 1) ? __cosf(ang) : __sinf(ang);
        }
        ueS[l * UE_STRIDE + k] = (_Float16)v;
    }
    // Zero padded rows 200..207 so WMMA tiles are clean.
    for (int i = tid; i < (LPAD - L_) * COND; i += NTHREADS) {
        const int l = L_ + i / COND, k = i % COND;
        ueS[l * UE_STRIDE + k] = (_Float16)0.0f;
    }
    __syncthreads();

    // -- mean over L (2-way split over row halves) ------------------------
    if (tid < 2 * COND) {
        const int c   = tid % COND;
        const int seg = tid / COND;
        float s = 0.0f;
        const int l0 = seg * LHALF;
        for (int l = l0; l < l0 + LHALF; ++l)
            s += (float)ueS[l * UE_STRIDE + c];
        partS[seg * COND + c] = s;
    }
    __syncthreads();
    if (tid < COND)
        meanS[tid] = (partS[tid] + partS[COND + tid]) * (1.0f / (float)L_);
    __syncthreads();

    // -- m1[c] = mean . w1[c,:] + b2[c] (2-way split over k halves) -------
    if (tid < 2 * COND) {
        const int c   = tid % COND;
        const int seg = tid / COND;
        float acc = (seg == 0) ? b2[c] : 0.0f;
        const int k0 = seg * (COND / 2);
        const float* w1row = w1 + (size_t)c * COND + k0;
        for (int k = 0; k < COND / 2; ++k) acc += meanS[k0 + k] * w1row[k];
        partS[seg * COND + c] = acc;
    }
    __syncthreads();
    if (tid < COND) m1S[tid] = partS[tid] + partS[COND + tid];
    __syncthreads();

    // -- WMMA: each wave owns one 16-row tile, sweeps all 10 col tiles ----
    const int wave = tid >> 5;            // == mtile, 0..12
    const int lane = tid & 31;
    const int half = lane >> 4;           // 0: lanes 0-15, 1: lanes 16-31
    const int lm   = lane & 15;
    const int mt   = wave;

    // A fragments: loaded once, reused across all 10 n-tiles.
    // Layout (16-bit A 16x32): VGPR 0-3: K = 8*half + 2i (+1);
    //                          VGPR 4-7: K = 16 + 8*half + 2i (+1).
    v16h afrag[KSTEPS];
    {
        const int abase = (mt * 16 + lm) * UE_STRIDE + 8 * half;
#pragma unroll
        for (int ks = 0; ks < KSTEPS; ++ks) {
            const int arow = abase + ks * 32;
#pragma unroll
            for (int i2 = 0; i2 < 4; ++i2) {
                afrag[ks][2 * i2]     = ueS[arow + 2 * i2];
                afrag[ks][2 * i2 + 1] = ueS[arow + 2 * i2 + 1];
                afrag[ks][2 * i2 + 8] = ueS[arow + 16 + 2 * i2];
                afrag[ks][2 * i2 + 9] = ueS[arow + 16 + 2 * i2 + 1];
            }
        }
    }

    float rowsum[8];
#pragma unroll
    for (int r = 0; r < 8; ++r) rowsum[r] = 0.0f;

    for (int nt = 0; nt < NTILES; ++nt) {
        v8f acc = {};
#pragma unroll
        for (int ks = 0; ks < KSTEPS; ++ks) {
            // B fragment (32x16 f16): lanes = N; VGPR 0-3: K = 8*half + 2i (+1),
            // VGPR 4-7: K = 16 + 8*half + 2i (+1). B[k][n] = w2[n][k].
            v16h bf;
            const int brow = (nt * 16 + lm) * W2_STRIDE + ks * 32 + 8 * half;
#pragma unroll
            for (int i2 = 0; i2 < 4; ++i2) {
                bf[2 * i2]     = w2S[brow + 2 * i2];
                bf[2 * i2 + 1] = w2S[brow + 2 * i2 + 1];
                bf[2 * i2 + 8] = w2S[brow + 16 + 2 * i2];
                bf[2 * i2 + 9] = w2S[brow + 16 + 2 * i2 + 1];
            }
            acc = __builtin_amdgcn_wmma_f32_16x16x32_f16(
                /*neg_a=*/false, afrag[ks], /*neg_b=*/false, bf,
                /*c_mod=*/(short)0, acc, /*reuse_a=*/false, /*reuse_b=*/false);
        }
        // Fused epilogue: fast sigmoid (v_rcp_f32) + weight, per-row regs.
        const int   c  = nt * 16 + lm;
        const float m1 = m1S[c];
        const float w  = wcS[c];
#pragma unroll
        for (int r = 0; r < 8; ++r) {
            const float p  = acc[r] + m1;
            const float sg = __builtin_amdgcn_rcpf(1.0f + __expf(-p));
            rowsum[r] += sg * w;
        }
    }

    // Butterfly-reduce rowsum over the 16 lanes of each half-wave; then one
    // plain store per row (each row owned by exactly one wave/lane).
#pragma unroll
    for (int r = 0; r < 8; ++r) {
        float v = rowsum[r];
        v += __shfl_xor(v, 1, 32);
        v += __shfl_xor(v, 2, 32);
        v += __shfl_xor(v, 4, 32);
        v += __shfl_xor(v, 8, 32);
        if (lm == 0) scoreS[mt * 16 + r + 8 * half] = v;  // rows>=200 unused
    }
    __syncthreads();

    // -- uemb[c] = sum_l score[l] * ue[l][c] (2-way split) ----------------
    if (tid < 2 * COND) {
        const int c   = tid % COND;
        const int seg = tid / COND;
        float u = 0.0f;
        const int l0 = seg * LHALF;
        for (int l = l0; l < l0 + LHALF; ++l)
            u += scoreS[l] * (float)ueS[l * UE_STRIDE + c];
        partS[seg * COND + c] = u;
    }
    __syncthreads();
    if (tid < COND) uembS[tid] = partS[tid] + partS[COND + tid];
    __syncthreads();

    // -- Broadcast user_info rows [160, 320) ------------------------------
    for (int i = tid; i < COND * EMB; i += NTHREADS) {
        const int c = i / EMB;
        const int e = i % EMB;
        out[(size_t)b * OUT_ROWS * EMB + (COND + c) * EMB + e] = uembS[c];
    }
}

// ---------------------------------------------------------------------------
extern "C" void kernel_launch(void* const* d_in, const int* in_sizes, int n_in,
                              void* d_out, int out_size, void* d_ws, size_t ws_size,
                              hipStream_t stream) {
    (void)in_sizes; (void)n_in; (void)out_size; (void)d_ws; (void)ws_size;

    const int*   tp    = (const int*)d_in[0];
    const int*   loc   = (const int*)d_in[1];
    const float* emb   = (const float*)d_in[2];
    const int*   ST    = (const int*)d_in[3];
    const float* lea   = (const float*)d_in[4];
    const float* noise = (const float*)d_in[5];
    const int*   t     = (const int*)d_in[6];
    const float* w1    = (const float*)d_in[7];
    const float* w2    = (const float*)d_in[8];
    const float* b2v   = (const float*)d_in[9];
    const float* wc    = (const float*)d_in[10];
    const float* ah    = (const float*)d_in[11];
    float*       out   = (float*)d_out;

    st_noisy_kernel<<<B_, 256, 0, stream>>>(tp, loc, emb, lea, noise, t, ah, out);

    const size_t smem = (size_t)(LPAD * UE_STRIDE + COND * W2_STRIDE) * sizeof(_Float16)
                      + (size_t)(LPAD + 6 * COND + 64) * sizeof(float);
    hipFuncSetAttribute(reinterpret_cast<const void*>(user_info_kernel),
                        hipFuncAttributeMaxDynamicSharedMemorySize, (int)smem);
    user_info_kernel<<<B_, NTHREADS, smem, stream>>>(ST, lea, w1, w2, b2v, wc, out);
}
